// LSTM_65266323030043
// MI455X (gfx1250) — matrix operand order repaired and checked by reading, not compiled, not run on hardware
//
#include <hip/hip_runtime.h>

// ---------------------------------------------------------------------------
// Types
// ---------------------------------------------------------------------------
typedef __bf16 bf16_t;
typedef __attribute__((ext_vector_type(8)))  __bf16 v8bf;
typedef __attribute__((ext_vector_type(16))) __bf16 v16bf;
typedef __attribute__((ext_vector_type(8)))  float  v8f;
typedef __attribute__((ext_vector_type(4)))  unsigned int v4u;
typedef __attribute__((ext_vector_type(8)))  int v8i;
typedef __attribute__((ext_vector_type(4)))  int v4i;

#define LSTM_B   1024
#define LSTM_M   64
#define LSTM_IN  512
#define LSTM_H   512
#define XROW     (LSTM_M * LSTM_IN)   // 32768 floats per batch row of x
#define KTOT     (LSTM_IN + LSTM_H)   // 1024

// Block tile: 32 batch rows x 64 H-columns (x4 gates). 128 threads = 4 wave32.
// Wave w owns columns n0 = 16*w and BOTH row subtiles (M=0 and M=16), so each
// B (weight) fragment feeds two back-to-back WMMAs.
#define TROWS    32
#define TCOLS    64
#define XSTAGE   256                  // K-stage width for X in LDS
#define LDX      (XSTAGE + 8)         // 264: row stride (bf16) -> banks shift 4/row
#define LDH      (LSTM_H + 8)         // 520: matches TDM pad (16B per 1024B row)

// ---------------------------------------------------------------------------
// float -> bf16 using the hardware convert (RTNE)
// ---------------------------------------------------------------------------
__device__ inline bf16_t f2bf(float f) { return static_cast<bf16_t>(f); }

__device__ inline float sigmoidf_fast(float v) {
  return 1.0f / (1.0f + __expf(-v));
}

// ---------------------------------------------------------------------------
// WMMA fragment loaders (CDNA5 wave32 layouts, CDNA5 ISA §7.12.2)
// A (16x32 bf16): lane&15 = row M; lanes 0-15 hold K [0..7]+[16..23],
//                 lanes 16-31 hold K [8..15]+[24..31].
// B stored transposed row-major [n][k]: lane&15 = col N; lanes 0-15 hold
//                 K [0..7]+[16..23], lanes 16-31 hold K [8..15]+[24..31].
// ---------------------------------------------------------------------------
__device__ inline v16bf load_a_bf16(const bf16_t* tile, int ld) {
  const int lane = threadIdx.x & 31;
  const bf16_t* p = tile + (size_t)(lane & 15) * ld + ((lane >> 4) << 3);
  v8bf lo = *(const v8bf*)(p);
  v8bf hi = *(const v8bf*)(p + 16);
  return __builtin_shufflevector(lo, hi, 0, 1, 2, 3, 4, 5, 6, 7,
                                         8, 9, 10, 11, 12, 13, 14, 15);
}

__device__ inline v16bf load_b_bf16t(const bf16_t* __restrict__ tileT, int ld) {
  const int lane = threadIdx.x & 31;
  const bf16_t* p = tileT + (size_t)(lane & 15) * ld + ((lane >> 4) << 3);
  v8bf lo = *(const v8bf*)(p);
  v8bf hi = *(const v8bf*)(p + 16);
  return __builtin_shufflevector(lo, hi, 0, 1, 2, 3, 4, 5, 6, 7,
                                         8, 9, 10, 11, 12, 13, 14, 15);
}

// ---------------------------------------------------------------------------
// Prep: Wt[g][n][k] bf16 (k<512 from Wx_g, else Wh_g), transposed.
// ---------------------------------------------------------------------------
__global__ void prep_weights(const float* __restrict__ Wx0, const float* __restrict__ Wx1,
                             const float* __restrict__ Wx2, const float* __restrict__ Wx3,
                             const float* __restrict__ Wh0, const float* __restrict__ Wh1,
                             const float* __restrict__ Wh2, const float* __restrict__ Wh3,
                             bf16_t* __restrict__ Wt) {
  int idx = blockIdx.x * blockDim.x + threadIdx.x;   // < 4*512*1024
  int g   = idx >> 19;
  int rem = idx & 0x7FFFF;
  int n   = rem >> 10;
  int k   = rem & 1023;
  const float* Wx = (g == 0) ? Wx0 : (g == 1) ? Wx1 : (g == 2) ? Wx2 : Wx3;
  const float* Wh = (g == 0) ? Wh0 : (g == 1) ? Wh1 : (g == 2) ? Wh2 : Wh3;
  float v = (k < LSTM_IN) ? Wx[(size_t)k * LSTM_H + n]
                          : Wh[(size_t)(k - LSTM_IN) * LSTM_H + n];
  Wt[idx] = f2bf(v);
}

__global__ void prep_state(float* __restrict__ C, bf16_t* __restrict__ H0,
                           float* __restrict__ bias,
                           const float* __restrict__ bx0, const float* __restrict__ bx1,
                           const float* __restrict__ bx2, const float* __restrict__ bx3,
                           const float* __restrict__ bh0, const float* __restrict__ bh1,
                           const float* __restrict__ bh2, const float* __restrict__ bh3) {
  int idx = blockIdx.x * blockDim.x + threadIdx.x;   // < 1024*512
  C[idx]  = 0.0f;
  H0[idx] = __builtin_bit_cast(bf16_t, (unsigned short)0);
  if (idx < 4 * LSTM_H) {
    int g = idx >> 9, n = idx & 511;
    const float* bx = (g == 0) ? bx0 : (g == 1) ? bx1 : (g == 2) ? bx2 : bx3;
    const float* bh = (g == 0) ? bh0 : (g == 1) ? bh1 : (g == 2) ? bh2 : bh3;
    bias[idx] = bx[n] + bh[n];
  }
}

// ---------------------------------------------------------------------------
// One LSTM time step.
// Grid: 32 batch tiles x 8 column tiles; 128 threads = 4 wave32.
// H tile is DMA'd into LDS by the Tensor Data Mover (row-padded),
// overlapped with the X-part GEMM (staged through LDS in 256-wide K stages).
// ---------------------------------------------------------------------------
__global__ __launch_bounds__(128) void lstm_step(
    const float* __restrict__ x, const bf16_t* __restrict__ Wt,
    const float* __restrict__ bias, const bf16_t* __restrict__ Hin,
    bf16_t* __restrict__ Hout, float* __restrict__ Cst,
    float* __restrict__ Hf32, int m) {
  __shared__ bf16_t sX[TROWS * LDX];   // 16,896 B
  __shared__ bf16_t sH[TROWS * LDH];   // 33,280 B

  const int tid   = threadIdx.x;
  const int wave  = tid >> 5;          // 0..3
  const int lane  = tid & 31;
  const int brow0 = blockIdx.x * TROWS;
  const int bcol0 = blockIdx.y * TCOLS;

  // ---- Issue TDM: Hin[brow0 .. brow0+31][0..511] (bf16) -> sH, padded ----
  // D# group0: count=1 | lds_addr | global_addr | type=2
  // D# group1: data_size=2B, pad_enable, pad_interval=256 DW (1024 B),
  //            pad_amount=4 DW (16 B) -> LDS row stride 520 elements.
  if (wave == 0) {
    unsigned long long ga =
        (unsigned long long)(uintptr_t)(Hin + (size_t)brow0 * LSTM_H);
    unsigned lds_off = (unsigned)(uintptr_t)(&sH[0]);
    v4u g0;
    g0[0] = 1u;                                   // count = 1 (valid D#)
    g0[1] = lds_off;                              // lds_addr (bytes)
    g0[2] = (unsigned)(ga & 0xFFFFFFFFu);         // global_addr[31:0]
    g0[3] = (unsigned)((ga >> 32) & 0x01FFFFFFu)  // global_addr[56:32]
            | 0x80000000u;                        // type = 2 ("image")
    v8i g1;
    g1[0] = (int)((1u << 16)      // data_size = 2 bytes
                | (1u << 20)      // pad_enable
                | (7u << 22)      // pad_interval = 256 DWORDs (1024 B)
                | (3u << 25));    // pad_amount   = 4 DWORDs (16 B)
    g1[1] = (int)(512u << 16);    // tensor_dim0 = 512
    g1[2] = (int)(1024u << 16);   // tensor_dim1 = 1024
    g1[3] = (int)(512u << 16);    // tile_dim0 = 512
    g1[4] = 32;                   // tile_dim1 = 32
    g1[5] = 512;                  // tensor_dim0_stride = 512
    g1[6] = 0;
    g1[7] = 0;
    v4i gz4 = {0, 0, 0, 0};
    v8i gz8 = {0, 0, 0, 0, 0, 0, 0, 0};
    __builtin_amdgcn_tensor_load_to_lds(g0, g1, gz4, gz4, gz8, 0);
  }

  // ---- columns owned by this wave; both row subtiles (M=0, M=16) ----
  const int n0 = wave * 16;
  v8f acc[2][4] = {};                  // [row subtile][gate i,f,o,c]

  // ---- X @ Wx (K = 0..511) in two LDS stages of 256 ----
  const float* xsrc = x + (size_t)brow0 * XROW + (size_t)m * LSTM_IN;
  const int lr = tid >> 2;             // 0..31 (row)
  const int lc = (tid & 3) * 64;       // 0..192 (col base, 64 per thread)
  for (int kb = 0; kb < LSTM_IN; kb += XSTAGE) {
    // cooperative convert f32 -> bf16 into sX
    const float* rp = xsrc + (size_t)lr * XROW + kb + lc;
    bf16_t* wp = sX + lr * LDX + lc;
#pragma unroll
    for (int j = 0; j < 64; j += 8) {
      float4 f0 = *(const float4*)(rp + j);
      float4 f1 = *(const float4*)(rp + j + 4);
      v8bf w;
      w[0] = f2bf(f0.x); w[1] = f2bf(f0.y); w[2] = f2bf(f0.z); w[3] = f2bf(f0.w);
      w[4] = f2bf(f1.x); w[5] = f2bf(f1.y); w[6] = f2bf(f1.z); w[7] = f2bf(f1.w);
      *(v8bf*)(wp + j) = w;
    }
    __syncthreads();

    for (int k0 = 0; k0 < XSTAGE; k0 += 32) {
      v16bf a0 = load_a_bf16(sX + 0  * LDX + k0, LDX);
      v16bf a1 = load_a_bf16(sX + 16 * LDX + k0, LDX);
#pragma unroll
      for (int g = 0; g < 4; ++g) {
        const bf16_t* bp =
            Wt + (size_t)(g * LSTM_H + bcol0 + n0) * KTOT + kb + k0;
        v16bf b = load_b_bf16t(bp, KTOT);
        acc[0][g] = __builtin_amdgcn_wmma_f32_16x16x32_bf16(
            false, a0, false, b, (short)0, acc[0][g], false, false);
        acc[1][g] = __builtin_amdgcn_wmma_f32_16x16x32_bf16(
            false, a1, false, b, (short)0, acc[1][g], false, false);
      }
    }
    __syncthreads();   // before next stage overwrites sX
  }

  // ---- wait for the TDM'd H tile, then H @ Wh (K = 512..1023) ----
  if (wave == 0) __builtin_amdgcn_s_wait_tensorcnt(0);
  __syncthreads();

  for (int k0 = 0; k0 < LSTM_H; k0 += 32) {
    v16bf a0 = load_a_bf16(sH + 0  * LDH + k0, LDH);
    v16bf a1 = load_a_bf16(sH + 16 * LDH + k0, LDH);
#pragma unroll
    for (int g = 0; g < 4; ++g) {
      const bf16_t* bp =
          Wt + (size_t)(g * LSTM_H + bcol0 + n0) * KTOT + LSTM_IN + k0;
      v16bf b = load_b_bf16t(bp, KTOT);
      acc[0][g] = __builtin_amdgcn_wmma_f32_16x16x32_bf16(
          false, a0, false, b, (short)0, acc[0][g], false, false);
      acc[1][g] = __builtin_amdgcn_wmma_f32_16x16x32_bf16(
          false, a1, false, b, (short)0, acc[1][g], false, false);
    }
  }

  // ---- gate nonlinearities + C/H update ----
  // C/D layout: VGPR r -> M = r + 8*(lane>=16), N = lane&15.
  const int n   = lane & 15;
  const int col = bcol0 + n0 + n;
  const float bi = bias[0 * LSTM_H + col];
  const float bf = bias[1 * LSTM_H + col];
  const float bo = bias[2 * LSTM_H + col];
  const float bc = bias[3 * LSTM_H + col];
  const int mbase = (lane >> 4) * 8;

#pragma unroll
  for (int si = 0; si < 2; ++si) {
    const int m0 = si * 16;
#pragma unroll
    for (int r = 0; r < 8; ++r) {
      const int row = brow0 + m0 + mbase + r;
      const size_t gi = (size_t)row * LSTM_H + col;
      const float ig = sigmoidf_fast(acc[si][0][r] + bi);
      const float fg = sigmoidf_fast(acc[si][1][r] + bf);
      const float og = sigmoidf_fast(acc[si][2][r] + bo);
      const float ct = tanhf(acc[si][3][r] + bc);
      const float cn = fg * Cst[gi] + ig * ct;
      Cst[gi] = cn;
      const float h = og * tanhf(cn);
      Hout[gi] = f2bf(h);                  // bf16 H for next step
      if (m == LSTM_M - 1) Hf32[gi] = h;   // final answer into d_out
    }
  }
}

// ---------------------------------------------------------------------------
// Launch
// ---------------------------------------------------------------------------
extern "C" void kernel_launch(void* const* d_in, const int* in_sizes, int n_in,
                              void* d_out, int out_size, void* d_ws, size_t ws_size,
                              hipStream_t stream) {
  (void)in_sizes; (void)n_in; (void)out_size; (void)ws_size;

  const float* x = (const float*)d_in[0];
  const float* Wx[4] = {(const float*)d_in[1],  (const float*)d_in[5],
                        (const float*)d_in[9],  (const float*)d_in[13]};
  const float* bx[4] = {(const float*)d_in[2],  (const float*)d_in[6],
                        (const float*)d_in[10], (const float*)d_in[14]};
  const float* Wh[4] = {(const float*)d_in[3],  (const float*)d_in[7],
                        (const float*)d_in[11], (const float*)d_in[15]};
  const float* bh[4] = {(const float*)d_in[4],  (const float*)d_in[8],
                        (const float*)d_in[12], (const float*)d_in[16]};

  char* ws = (char*)d_ws;
  bf16_t* Wt   = (bf16_t*)(ws);                                     // 4 MB
  float*  bias = (float*) (ws + ((size_t)4 << 20));                 // 8 KB
  bf16_t* H0   = (bf16_t*)(ws + ((size_t)4 << 20) + (64u << 10));   // 1 MB
  bf16_t* H1   = (bf16_t*)(ws + ((size_t)5 << 20) + (64u << 10));   // 1 MB
  float*  C    = (float*) (ws + ((size_t)6 << 20) + (64u << 10));   // 2 MB
  float*  Hf32 = (float*)d_out;                                     // (1024,512)

  prep_weights<<<(4 * LSTM_H * KTOT) / 256, 256, 0, stream>>>(
      Wx[0], Wx[1], Wx[2], Wx[3], Wh[0], Wh[1], Wh[2], Wh[3], Wt);
  prep_state<<<(LSTM_B * LSTM_H) / 256, 256, 0, stream>>>(
      C, H0, bias, bx[0], bx[1], bx[2], bx[3], bh[0], bh[1], bh[2], bh[3]);

  dim3 grid(LSTM_B / TROWS, LSTM_H / TCOLS);   // 32 x 8 = 256 blocks
  dim3 blk(128);                               // 4 wave32
  for (int t = 0; t < LSTM_M; ++t) {
    const bf16_t* Hin = (t & 1) ? H1 : H0;
    bf16_t*       Ho  = (t & 1) ? H0 : H1;
    lstm_step<<<grid, blk, 0, stream>>>(x, Wt, bias, Hin, Ho, C, Hf32, t);
  }
}
